// Top2Router_3959959847165
// MI455X (gfx1250) — matrix operand order repaired and checked by reading, not compiled, not run on hardware
//
#include <hip/hip_runtime.h>

// ---------------------------------------------------------------------------
// Top-2 MoE router for MI455X (gfx1250), wave32 + WMMA bf16 path.
//   dispatch/combine: [16384, 64] fp32, expert_counts: [64] fp32
//   d_out layout: dispatch | combine | counts
//
// Roofline: x read = 268 MB -> ~11.5 us floor @ 23.3 TB/s. GEMM = 8.6 GFLOP,
// needs ~750 TFLOPS to stay bandwidth-bound -> bf16 WMMA (fp32 accum).
// 16 tokens per wave -> 1024 waves (128 blocks x 8 waves) to keep enough
// SIMDs busy to actually draw full HBM bandwidth; W (bf16, 512 KB) streams
// from L2 per wave (~45 TB/s aggregate L2, fine).
// ---------------------------------------------------------------------------

typedef __attribute__((ext_vector_type(16))) __bf16 v16bf;
typedef __attribute__((ext_vector_type(8)))  float  v8f;
typedef __attribute__((ext_vector_type(4)))  float  fvec4;

constexpr int kD   = 4096;       // d_model
constexpr int kE   = 64;         // num experts
constexpr int kTok = 4 * 4096;   // total tokens

// ---- prep: convert W to bf16 workspace; zero the counts accumulator -------
__global__ __launch_bounds__(256) void prep_kernel(const float* __restrict__ W,
                                                   __bf16* __restrict__ Wbf,
                                                   float* __restrict__ counts) {
  const int gid = blockIdx.x * blockDim.x + threadIdx.x;
  if (gid < kE * kD) Wbf[gid] = (__bf16)W[gid];
  if (gid < kE) counts[gid] = 0.0f;
}

// ---- main: logits GEMM (bf16 WMMA) + fused softmax/top2/scatter -----------
__global__ __launch_bounds__(256) void router_kernel(const float* __restrict__ x,
                                                     const __bf16* __restrict__ Wbf,
                                                     const float* __restrict__ bias,
                                                     float* __restrict__ dispatch,
                                                     float* __restrict__ combine,
                                                     float* __restrict__ counts) {
  __shared__ float scnt[kE];

  const int lane = threadIdx.x & 31;
  const int wave = (blockIdx.x * blockDim.x + threadIdx.x) >> 5;
  const int tok_base = wave * 16;           // 16 tokens per wave (1 M-tile)
  const int n    = lane & 15;               // column / expert-within-tile
  const int half = lane >> 4;               // wave half selects K sub-range

  if (threadIdx.x < kE) scnt[threadIdx.x] = 0.0f;
  __syncthreads();

  // accumulators: c[t] = 16x16 fp32 tile, experts 16t..16t+15
  v8f c[4];
#pragma unroll
  for (int t = 0; t < 4; ++t)
    c[t] = (v8f){0.f, 0.f, 0.f, 0.f, 0.f, 0.f, 0.f, 0.f};

  // A fragment base: lane (and lane+16) hold row m = tok_base + n.
  // A k order used here (contiguous, same as B): lanes 0-15 K = 0..15, lanes 16-31 K = 16..31.
  const float* a0 = x + (size_t)(tok_base + n) * kD + half * 16;

  // B fragment base: lane holds column (expert) 16t+n; lanes 0-15 K=0..15, 16-31 K=16..31.
  const __bf16* bp0 = Wbf + (size_t)(0 * 16 + n) * kD + half * 16;
  const __bf16* bp1 = Wbf + (size_t)(1 * 16 + n) * kD + half * 16;
  const __bf16* bp2 = Wbf + (size_t)(2 * 16 + n) * kD + half * 16;
  const __bf16* bp3 = Wbf + (size_t)(3 * 16 + n) * kD + half * 16;

  for (int k0 = 0; k0 < kD; k0 += 32) {
    v16bf bt[4];
    bt[0] = *(const v16bf*)(bp0 + k0);
    bt[1] = *(const v16bf*)(bp1 + k0);
    bt[2] = *(const v16bf*)(bp2 + k0);
    bt[3] = *(const v16bf*)(bp3 + k0);

    const fvec4 f0 = *(const fvec4*)(a0 + k0);
    const fvec4 f1 = *(const fvec4*)(a0 + k0 + 4);
    const fvec4 f2 = *(const fvec4*)(a0 + k0 + 8);
    const fvec4 f3 = *(const fvec4*)(a0 + k0 + 12);
    v16bf a;
#pragma unroll
    for (int i = 0; i < 4; ++i) {
      a[i]      = (__bf16)f0[i];
      a[4 + i]  = (__bf16)f1[i];
      a[8 + i]  = (__bf16)f2[i];
      a[12 + i] = (__bf16)f3[i];
    }
#pragma unroll
    for (int t = 0; t < 4; ++t)
      c[t] = __builtin_amdgcn_wmma_f32_16x16x32_bf16(
          /*neg_a=*/false, a, /*neg_b=*/false, bt[t],
          /*c_mod=*/(short)0, c[t], /*reuse_a=*/false, /*reuse_b=*/false);
  }

  // ---- fused epilogue on the WMMA C layout --------------------------------
  // C tile: VGPR v, lanes 0-15 -> token M=v; lanes 16-31 -> token M=8+v; N = lane&15.
  float bv[4];
#pragma unroll
  for (int t = 0; t < 4; ++t) bv[t] = bias[16 * t + n];

  float cacc[4] = {0.f, 0.f, 0.f, 0.f};
  const float kNegInf = -3.402823466e38f;

#pragma unroll
  for (int v = 0; v < 8; ++v) {
    const int tok = tok_base + half * 8 + v;
    float val[4];
#pragma unroll
    for (int t = 0; t < 4; ++t) val[t] = c[t][v] + bv[t];

    // top-1 (value + expert index), butterfly over the 16-lane half
    float m1 = val[0]; int i1 = n;
#pragma unroll
    for (int t = 1; t < 4; ++t) {
      const int e = 16 * t + n;
      if (val[t] > m1) { m1 = val[t]; i1 = e; }
    }
#pragma unroll
    for (int msk = 1; msk <= 8; msk <<= 1) {
      const float om = __shfl_xor(m1, msk, 32);
      const int   oi = __shfl_xor(i1, msk, 32);
      if (om > m1 || (om == m1 && oi < i1)) { m1 = om; i1 = oi; }
    }

    // top-2: exclude i1, reduce again
    float m2 = kNegInf; int i2 = 0x7fffffff;
#pragma unroll
    for (int t = 0; t < 4; ++t) {
      const int e = 16 * t + n;
      if (e != i1 && (val[t] > m2 || (val[t] == m2 && e < i2))) { m2 = val[t]; i2 = e; }
    }
#pragma unroll
    for (int msk = 1; msk <= 8; msk <<= 1) {
      const float om = __shfl_xor(m2, msk, 32);
      const int   oi = __shfl_xor(i2, msk, 32);
      if (om > m2 || (om == m2 && oi < i2)) { m2 = om; i2 = oi; }
    }

    // softmax denominator over all 64 experts
    float p[4];
    float s = 0.f;
#pragma unroll
    for (int t = 0; t < 4; ++t) { p[t] = __expf(val[t] - m1); s += p[t]; }
#pragma unroll
    for (int msk = 1; msk <= 8; msk <<= 1) s += __shfl_xor(s, msk, 32);
    const float inv = 1.0f / s;

    // scatter dense dispatch/combine (top-2 probs, zero elsewhere)
#pragma unroll
    for (int t = 0; t < 4; ++t) {
      const int e = 16 * t + n;
      const float o = (e == i1 || e == i2) ? p[t] * inv : 0.0f;
      const size_t off = (size_t)tok * kE + e;
      dispatch[off] = o;
      combine[off]  = o;
      cacc[t] += o;
    }
  }

  // expert_counts: fold wave halves, then LDS-reduce per block, 1 atomic/expert/block
#pragma unroll
  for (int t = 0; t < 4; ++t) cacc[t] += __shfl_xor(cacc[t], 16, 32);
  if (half == 0) {
#pragma unroll
    for (int t = 0; t < 4; ++t) atomicAdd(&scnt[16 * t + n], cacc[t]);
  }
  __syncthreads();
  if (threadIdx.x < kE) atomicAdd(&counts[threadIdx.x], scnt[threadIdx.x]);
}

// ---------------------------------------------------------------------------
extern "C" void kernel_launch(void* const* d_in, const int* in_sizes, int n_in,
                              void* d_out, int out_size, void* d_ws, size_t ws_size,
                              hipStream_t stream) {
  (void)in_sizes; (void)n_in; (void)out_size; (void)ws_size;
  const float* x = (const float*)d_in[0];   // [4,4096,4096]
  const float* W = (const float*)d_in[1];   // [64,4096]
  const float* b = (const float*)d_in[2];   // [64]

  float* out      = (float*)d_out;
  float* dispatch = out;
  float* combine  = out + (size_t)kTok * kE;
  float* counts   = out + 2 * (size_t)kTok * kE;

  __bf16* Wbf = (__bf16*)d_ws;              // 64*4096*2 = 512 KB scratch

  prep_kernel<<<(kE * kD + 255) / 256, 256, 0, stream>>>(W, Wbf, counts);

  // 16384 tokens / 16 per wave = 1024 waves; 8 waves (256 threads) per block.
  router_kernel<<<128, 256, 0, stream>>>(x, Wbf, b, dispatch, combine, counts);
}
